// selfAttention_19911468384688
// MI455X (gfx1250) — matrix operand order repaired and checked by reading, compiled, not run on hardware
//
#include <hip/hip_runtime.h>

// ---------------------------------------------------------------------------
// Fused self-attention (B=2, S=2048, D=1024, H=16, dk=64) for gfx1250.
// All matmuls use V_WMMA_F32_16X16X32_F16 (f16 A/B, f32 accumulate).
// ---------------------------------------------------------------------------

#define BB   2
#define SEQ  2048
#define DIM  1024
#define NH   16
#define DK   64

typedef _Float16 v16h __attribute__((ext_vector_type(16)));
typedef _Float16 v8h  __attribute__((ext_vector_type(8)));
typedef float    v8f  __attribute__((ext_vector_type(8)));

static __device__ __forceinline__ v8f wmma16(v16h a, v16h b, v8f c) {
  // (neg_a, A, neg_b, B, c_mod, C, reuse_a, reuse_b)
  return __builtin_amdgcn_wmma_f32_16x16x32_f16(false, a, false, b, (short)0, c,
                                                false, false);
}

// A-fragment (16x32 f16, M x K): lane&15 = row; lanes 0-15 hold K k0..k0+7 and
// k0+16..k0+23, lanes 16-31 hold K k0+8..k0+15 and k0+24..k0+31.
static __device__ __forceinline__ v16h load_a_frag(const _Float16* __restrict__ base,
                                                   int lda, int k0, int lane) {
  int row  = lane & 15;
  int koff = k0 + ((lane >> 4) ? 8 : 0);
  const _Float16* p = base + (size_t)row * lda + koff;
  v8h lo = *(const v8h*)p;
  v8h hi = *(const v8h*)(p + 16);
  v16h a;
#pragma unroll
  for (int j = 0; j < 8; ++j) { a[j] = lo[j]; a[j + 8] = hi[j]; }
  return a;
}

// B-fragment (32x16 f16, K x N), source stored K-contiguous per column
// (i.e. B^T row-major): lane&15 = column; lanes 0-15 hold K k0..k0+7 and
// k0+16..k0+23, lanes 16-31 hold K k0+8..k0+15 and k0+24..k0+31.
static __device__ __forceinline__ v16h load_b_frag(const _Float16* __restrict__ colbase,
                                                   int ldb, int k0, int lane) {
  int col = lane & 15;
  int kb  = k0 + ((lane >> 4) ? 8 : 0);
  const _Float16* p = colbase + (size_t)col * ldb + kb;
  v8h lo = *(const v8h*)p;
  v8h hi = *(const v8h*)(p + 16);
  v16h b;
#pragma unroll
  for (int j = 0; j < 8; ++j) { b[j] = lo[j]; b[j + 8] = hi[j]; }
  return b;
}

// ---------------------------------------------------------------------------
// K0a: f32 -> f16 cast
__global__ __launch_bounds__(256) void k_cast_h(const float* __restrict__ in,
                                                _Float16* __restrict__ out, int n) {
  int i = blockIdx.x * 256 + threadIdx.x;
  if (i < n) out[i] = (_Float16)in[i];
}

// K0b: Wt[n][k] = (f16) W[k][n]  (1024x1024)
__global__ __launch_bounds__(256) void k_transpose_cast(const float* __restrict__ W,
                                                        _Float16* __restrict__ Wt) {
  int idx = blockIdx.x * 256 + threadIdx.x;   // idx = n*1024 + k
  int n = idx >> 10, k = idx & 1023;
  Wt[idx] = (_Float16)W[(size_t)k * DIM + n];
}

// ---------------------------------------------------------------------------
// WMMA GEMM: C[M][N] (f32) = A[M][K] (f16 row-major) x Bt[N][K] (f16, B^T).
// N,K compile-time so the C-store epilogue folds row offsets into immediate
// store offsets instead of 64-bit address chains.
// Wave tile 32x64 (2x4 accumulators); block = 8 waves = 64x256.
template <int N, int K>
__global__ __launch_bounds__(256) void k_gemm_f16f32(const _Float16* __restrict__ A,
                                                     const _Float16* __restrict__ Bt,
                                                     float* __restrict__ C) {
  int w    = threadIdx.x >> 5;
  int lane = threadIdx.x & 31;
  int m0 = blockIdx.x * 64  + (w >> 2) * 32;
  int n0 = blockIdx.y * 256 + (w & 3)  * 64;
  const v8f vzero = {0.f, 0.f, 0.f, 0.f, 0.f, 0.f, 0.f, 0.f};
  v8f acc[2][4];
#pragma unroll
  for (int i = 0; i < 2; ++i)
#pragma unroll
    for (int j = 0; j < 4; ++j) acc[i][j] = vzero;

  const _Float16* arow0 = A + (size_t)(m0 + (lane & 15)) * K;
  const _Float16* arow1 = A + (size_t)(m0 + 16 + (lane & 15)) * K;
  const _Float16* bcol  = Bt + (size_t)(n0 + (lane & 15)) * K;

  for (int k0 = 0; k0 < K; k0 += 32) {
    v16h af[2];
    af[0] = load_a_frag(A + (size_t)m0 * K,        K, k0, lane);
    af[1] = load_a_frag(A + (size_t)(m0 + 16) * K, K, k0, lane);
    v16h bf[4];
#pragma unroll
    for (int j = 0; j < 4; ++j)
      bf[j] = load_b_frag(Bt + (size_t)(n0 + j * 16) * K, K, k0, lane);

    // gfx1250 prefetch path: pull next k-step's lines toward the WGP.
    if (k0 + 32 < K) {
      __builtin_prefetch(arow0 + k0 + 32, 0, 3);
      __builtin_prefetch(arow1 + k0 + 32, 0, 3);
      __builtin_prefetch(bcol + k0 + 32, 0, 3);
    }
#pragma unroll
    for (int i = 0; i < 2; ++i)
#pragma unroll
      for (int j = 0; j < 4; ++j) acc[i][j] = wmma16(af[i], bf[j], acc[i][j]);
  }

  int n = lane & 15, mb = (lane >> 4) * 8;
#pragma unroll
  for (int i = 0; i < 2; ++i)
#pragma unroll
    for (int j = 0; j < 4; ++j) {
      float* cb = C + (size_t)(m0 + i * 16 + mb) * N + (n0 + j * 16 + n);
#pragma unroll
      for (int r = 0; r < 8; ++r) cb[r * N] = acc[i][j][r];
    }
}

// ---------------------------------------------------------------------------
// K2: RoPE on q,k + pack to [B,H,S,dk] f16; v -> transposed [B,H,dk,S] f16.
__global__ __launch_bounds__(256) void k_rope_pack(const float* __restrict__ qf,
                                                   const float* __restrict__ kf,
                                                   const float* __restrict__ vf,
                                                   const float* __restrict__ fcos,
                                                   const float* __restrict__ fsin,
                                                   _Float16* __restrict__ qh,
                                                   _Float16* __restrict__ kh,
                                                   _Float16* __restrict__ vt) {
  int idx = blockIdx.x * 256 + threadIdx.x;
  int i = idx & 31;
  int h = (idx >> 5) & 15;
  int s = (idx >> 9) & 2047;
  int b = idx >> 20;
  size_t in = ((size_t)(b * SEQ + s)) * DIM + h * DK + 2 * i;
  float c  = fcos[s * 32 + i];
  float sn = fsin[s * 32 + i];
  size_t outq = (((size_t)(b * NH + h)) * SEQ + s) * DK + 2 * i;

  float re = qf[in], im = qf[in + 1];
  qh[outq]     = (_Float16)(re * c - im * sn);
  qh[outq + 1] = (_Float16)(re * sn + im * c);

  re = kf[in]; im = kf[in + 1];
  kh[outq]     = (_Float16)(re * c - im * sn);
  kh[outq + 1] = (_Float16)(re * sn + im * c);

  size_t vo = (((size_t)(b * NH + h)) * DK + 2 * i) * SEQ + s;
  vt[vo]       = (_Float16)vf[in];
  vt[vo + SEQ] = (_Float16)vf[in + 1];
}

// ---------------------------------------------------------------------------
// K3: fused attention. grid = (B*H, S/16), block = 256 (8 waves).
// LDS: scores 16x2052 f32 + 8x16x64 partials + 256 reduce scratch + p16
// (16x2056 f16) = 230,912 B of the 320 KB WGP budget (1 WG/WGP).
#define SCP  2052
#define P16S 2056
__global__ __launch_bounds__(256) void k_attn(const _Float16* __restrict__ qh,
                                              const _Float16* __restrict__ kh,
                                              const _Float16* __restrict__ vt,
                                              _Float16* __restrict__ attnh) {
  extern __shared__ float smem[];
  float* sc   = smem;                 // [16][SCP]
  float* part = sc + 16 * SCP;        // [8][16][64]
  float* red  = part + 8 * 16 * 64;   // [16][16]
  _Float16* p16 = (_Float16*)(red + 256);  // [16][P16S], 16B-aligned

  int bh = blockIdx.x, qb = blockIdx.y;
  int w = threadIdx.x >> 5, lane = threadIdx.x & 31;
  const _Float16* qbase = qh + ((size_t)bh * SEQ + qb * 16) * DK;
  const _Float16* kbase = kh + (size_t)bh * SEQ * DK;
  const _Float16* vbase = vt + (size_t)bh * DK * SEQ;
  const v8f vzero = {0.f, 0.f, 0.f, 0.f, 0.f, 0.f, 0.f, 0.f};
  int n = lane & 15, mb = (lane >> 4) * 8;

  // ---- Phase 1: scores(16 x 2048) = q(16x64) . k^T, scaled by 1/sqrt(64)
  v16h a0 = load_a_frag(qbase, DK, 0, lane);
  v16h a1 = load_a_frag(qbase, DK, 32, lane);
  int nbase = w * 256;                // each wave owns 256 key columns
  for (int nt = 0; nt < 16; ++nt) {
    const _Float16* kc = kbase + (size_t)(nbase + nt * 16) * DK;
    v16h b0 = load_b_frag(kc, DK, 0, lane);
    v16h b1 = load_b_frag(kc, DK, 32, lane);
    v8f c = vzero;
    c = wmma16(a0, b0, c);
    c = wmma16(a1, b1, c);
    int colg = nbase + nt * 16 + n;
#pragma unroll
    for (int r = 0; r < 8; ++r) sc[(mb + r) * SCP + colg] = c[r] * 0.125f;
  }
  __syncthreads();

  // ---- Phase 2: row softmax, applied twice (reference does softmax(softmax)).
  // Second pass writes the normalized probabilities as f16 into p16 so phase 3
  // can build A-fragments with two ds_load_b128 per lane.
  int row = threadIdx.x >> 4;
  int l16 = threadIdx.x & 15;
#pragma unroll 1
  for (int pass = 0; pass < 2; ++pass) {
    float m = -3.0e38f;
    for (int c = l16; c < SEQ; c += 16) m = fmaxf(m, sc[row * SCP + c]);
    red[row * 16 + l16] = m;
    __syncthreads();
    float rm = red[row * 16];
#pragma unroll
    for (int i = 1; i < 16; ++i) rm = fmaxf(rm, red[row * 16 + i]);
    __syncthreads();
    float s = 0.f;
    for (int c = l16; c < SEQ; c += 16) {
      float e = __expf(sc[row * SCP + c] - rm);
      sc[row * SCP + c] = e;
      s += e;
    }
    red[row * 16 + l16] = s;
    __syncthreads();
    float Z = 0.f;
#pragma unroll
    for (int i = 0; i < 16; ++i) Z += red[row * 16 + i];
    float inv = 1.0f / Z;
    __syncthreads();
    if (pass == 0) {
      for (int c = l16; c < SEQ; c += 16) sc[row * SCP + c] *= inv;
    } else {
      for (int c = l16; c < SEQ; c += 16)
        p16[row * P16S + c] = (_Float16)(sc[row * SCP + c] * inv);
    }
    __syncthreads();
  }

  // ---- Phase 3: out(16x64) = p(16x2048) . v(2048x64); waves split K=2048
  v8f acc[4];
#pragma unroll
  for (int j = 0; j < 4; ++j) acc[j] = vzero;
  int kc0 = w * 256;
  for (int kk = 0; kk < 256; kk += 32) {
    int kb = kc0 + kk;
    int arow = lane & 15;
    int ak = kb + ((lane >> 4) ? 8 : 0);
    const _Float16* prow = p16 + (size_t)arow * P16S + ak;
    v8h lo = *(const v8h*)prow;
    v8h hi = *(const v8h*)(prow + 16);
    v16h a;
#pragma unroll
    for (int j = 0; j < 8; ++j) { a[j] = lo[j]; a[j + 8] = hi[j]; }
#pragma unroll
    for (int nt = 0; nt < 4; ++nt) {
      v16h b = load_b_frag(vbase + (size_t)(nt * 16) * SEQ, SEQ, kb, lane);
      acc[nt] = wmma16(a, b, acc[nt]);
    }
  }
#pragma unroll
  for (int nt = 0; nt < 4; ++nt)
#pragma unroll
    for (int r = 0; r < 8; ++r)
      part[(w * 16 + mb + r) * 64 + nt * 16 + n] = acc[nt][r];
  __syncthreads();

  int bb = bh >> 4, hh = bh & 15;
  for (int e = threadIdx.x; e < 1024; e += 256) {
    int m2 = e >> 6, d = e & 63;
    float sum = 0.f;
#pragma unroll
    for (int ww = 0; ww < 8; ++ww) sum += part[(ww * 16 + m2) * 64 + d];
    int sidx = qb * 16 + m2;
    attnh[((size_t)bb * SEQ + sidx) * DIM + hh * DK + d] = (_Float16)sum;
  }
}

// ---------------------------------------------------------------------------
extern "C" void kernel_launch(void* const* d_in, const int* in_sizes, int n_in,
                              void* d_out, int out_size, void* d_ws, size_t ws_size,
                              hipStream_t stream) {
  (void)in_sizes; (void)n_in; (void)out_size; (void)ws_size;
  const float* x    = (const float*)d_in[0];
  const float* fcos = (const float*)d_in[1];
  const float* fsin = (const float*)d_in[2];
  const float* Wq   = (const float*)d_in[3];
  const float* Wk   = (const float*)d_in[4];
  const float* Wv   = (const float*)d_in[5];
  const float* Wo   = (const float*)d_in[6];
  float* out = (float*)d_out;

  const size_t MSD = (size_t)BB * SEQ * DIM;   // 4M elements
  char* ws = (char*)d_ws;
  size_t off = 0;
  auto alloc = [&](size_t bytes) -> char* {
    char* p = ws + off;
    off = (off + bytes + 255) & ~(size_t)255;
    return p;
  };
  _Float16* xh    = (_Float16*)alloc(MSD * 2);
  _Float16* wtq   = (_Float16*)alloc((size_t)DIM * DIM * 2);
  _Float16* wtk   = (_Float16*)alloc((size_t)DIM * DIM * 2);
  _Float16* wtv   = (_Float16*)alloc((size_t)DIM * DIM * 2);
  _Float16* wto   = (_Float16*)alloc((size_t)DIM * DIM * 2);
  float*    qf    = (float*)alloc(MSD * 4);
  float*    kf    = (float*)alloc(MSD * 4);
  float*    vf    = (float*)alloc(MSD * 4);
  _Float16* qhh   = (_Float16*)alloc(MSD * 2);
  _Float16* khh   = (_Float16*)alloc(MSD * 2);
  _Float16* vtt   = (_Float16*)alloc(MSD * 2);
  _Float16* attnh = (_Float16*)alloc(MSD * 2);

  // K0: casts
  k_cast_h<<<(int)(MSD / 256), 256, 0, stream>>>(x, xh, (int)MSD);
  k_transpose_cast<<<4096, 256, 0, stream>>>(Wq, wtq);
  k_transpose_cast<<<4096, 256, 0, stream>>>(Wk, wtk);
  k_transpose_cast<<<4096, 256, 0, stream>>>(Wv, wtv);
  k_transpose_cast<<<4096, 256, 0, stream>>>(Wo, wto);

  // K1: q/k/v projections (M=4096, N=1024, K=1024)
  dim3 ggrid(BB * SEQ / 64, DIM / 256);
  k_gemm_f16f32<DIM, DIM><<<ggrid, 256, 0, stream>>>(xh, wtq, qf);
  k_gemm_f16f32<DIM, DIM><<<ggrid, 256, 0, stream>>>(xh, wtk, kf);
  k_gemm_f16f32<DIM, DIM><<<ggrid, 256, 0, stream>>>(xh, wtv, vf);

  // K2: RoPE + layout pack
  k_rope_pack<<<8192, 256, 0, stream>>>(qf, kf, vf, fcos, fsin, qhh, khh, vtt);

  // K3: fused attention (double softmax)
  const int smemBytes = (16 * SCP + 8 * 16 * 64 + 256) * 4 + 16 * P16S * 2;
  k_attn<<<dim3(BB * NH, SEQ / 16), 256, smemBytes, stream>>>(qhh, khh, vtt, attnh);

  // K4: output projection straight into d_out
  k_gemm_f16f32<DIM, DIM><<<ggrid, 256, 0, stream>>>(attnh, wto, out);
}
